// UNet_25005299597349
// MI455X (gfx1250) — matrix operand order repaired and checked by reading, not compiled, not run on hardware
//
#include <hip/hip_runtime.h>
#include <cstdint>
#include <cstddef>

// ---------------------------------------------------------------------------
// Types for CDNA5 WMMA
// ---------------------------------------------------------------------------
typedef __attribute__((ext_vector_type(8)))  float          v8f;
typedef __attribute__((ext_vector_type(16))) __bf16         v16bf;
typedef __attribute__((ext_vector_type(16))) unsigned short v16us;

#if defined(__gfx1250__) && __has_builtin(__builtin_amdgcn_tensor_load_to_lds)
#define USE_TDM 1
typedef __attribute__((ext_vector_type(4))) unsigned int v4u;
typedef __attribute__((ext_vector_type(8))) int          v8i;
typedef __attribute__((ext_vector_type(4))) int          v4i;
#endif

__device__ __forceinline__ unsigned short f2bf(float f) {
  unsigned int u = __builtin_bit_cast(unsigned int, f);
  u += 0x7fffu + ((u >> 16) & 1u);            // round-to-nearest-even
  return (unsigned short)(u >> 16);
}

// GEMM tiling: block = 128 threads (4 waves); tile 32(M) x 128(N); K chunk 32.
#define TNB 128
#define TKC 32

// ---------------------------------------------------------------------------
// Weights f32 [Cout][K] -> bf16 packed A fragments [Cout/16][nch][32 lanes][16]
// Lane half h holds k = 16h..16h+15 of the A(16x32) operand (same k order as B), so the GEMM
// loads each lane's fragment as 32 contiguous bytes (2 x b128).
// ---------------------------------------------------------------------------
__global__ void wcvt_pack_kernel(const float* __restrict__ w, unsigned short* __restrict__ wb,
                                 int Cout, int K, int nch) {
  long total = (long)(Cout >> 4) * nch * 32 * 16;
  for (long i = (long)blockIdx.x * blockDim.x + threadIdx.x; i < total;
       i += (long)gridDim.x * blockDim.x) {
    int  h    = (int)(i & 15);
    int  lane = (int)((i >> 4) & 31);
    long t    = i >> 9;                 // mt*nch + c
    int  c    = (int)(t % nch);
    int  mt   = (int)(t / nch);
    bool lo   = lane < 16;
    int  m    = lane & 15;
    int  ka   = (lo ? 0 : 16) + h;
    int  k    = c * 32 + ka;
    int  co   = mt * 16 + m;
    float v = (k < K) ? w[(long)co * K + k] : 0.0f;
    wb[i] = f2bf(v);
  }
}

// ---------------------------------------------------------------------------
// im2col packed: x f32 [Cin][Hin][Hin] -> col bf16 [nch][Npix][32]
// (k fastest within a 32-chunk -> B fragment for pixel n is 32 contiguous
//  ushorts; a block's 32x128 B tile is one contiguous 8KB region for the TDM)
// ---------------------------------------------------------------------------
__global__ void im2col_pack_kernel(const float* __restrict__ x, unsigned short* __restrict__ col,
                                   int Cin, int Hin, int Ho, int stride, int ks, int pad,
                                   int nch, int Npix) {
  long total = (long)nch * Npix * 32;
  int  kk2   = ks * ks;
  for (long i = (long)blockIdx.x * blockDim.x + threadIdx.x; i < total;
       i += (long)gridDim.x * blockDim.x) {
    int  kk = (int)(i & 31);
    long t  = i >> 5;
    int  p  = (int)(t % Npix);
    int  c  = (int)(t / Npix);
    int  k  = c * 32 + kk;
    float v = 0.0f;
    if (k < Cin * kk2) {
      int ci  = k / kk2;
      int off = k % kk2;
      int ky = off / ks, kx = off % ks;
      int yo = p / Ho,  xo = p % Ho;
      int yy = yo * stride + ky - pad;
      int xx = xo * stride + kx - pad;
      if (yy >= 0 && yy < Hin && xx >= 0 && xx < Hin)
        v = x[((long)ci * Hin + yy) * Hin + xx];
    }
    col[i] = f2bf(v);
  }
}

// ---------------------------------------------------------------------------
// TDM issue: 1D contiguous copy of one 8KB B tile (2048 dwords) into LDS
// ---------------------------------------------------------------------------
#if USE_TDM
__device__ __forceinline__ void tdm_issue_b(const unsigned short* gsrc, unsigned int ldsoff) {
  unsigned long long ga = (unsigned long long)(uintptr_t)gsrc;
  // D# group0: count=1 | lds_addr | global_addr[56:0] | type=2
  v4u g0 = { 1u,
             ldsoff,
             (unsigned int)(ga & 0xffffffffu),
             (unsigned int)((ga >> 32) & 0x01ffffffu) | (2u << 30) };
  // D# group1: data_size=4B; tensor_dim0 = tile_dim0 = 2048 dwords;
  //            tensor_dim1 = tile_dim1 = 1 row; dim0_stride = 2048
  v8i g1 = { (int)(2u << 16),
             (int)((2048u & 0xffffu) << 16),
             (int)((2048u >> 16) | (1u << 16)),
             (int)(2048u << 16),
             (int)1,
             (int)2048,
             0, 0 };
  v4i gz = {0, 0, 0, 0};
#if __clang_major__ >= 23
  v8i gz8 = {0, 0, 0, 0, 0, 0, 0, 0};
  __builtin_amdgcn_tensor_load_to_lds(g0, g1, gz, gz, gz8, 0);
#else
  __builtin_amdgcn_tensor_load_to_lds(g0, g1, gz, gz, 0);
#endif
}
#endif

// ---------------------------------------------------------------------------
// WMMA GEMM:  C[Cout][Npix] = A(packed bf16) * B(packed bf16) + bias
// Double-buffered LDS; next B tile streamed by the Tensor Data Mover while the
// current tile is consumed by 4 WMMAs per wave per chunk (2 A frag x 2 B frag).
// ---------------------------------------------------------------------------
__global__ __launch_bounds__(128) void gemm_wmma_bias(
    const unsigned short* __restrict__ Apack,   // [Cout/16][nch][32][16]
    const unsigned short* __restrict__ Bpack,   // [nch][Npix][32]
    const float* __restrict__ bias,
    float* __restrict__ C,
    int Cout, int nch, int Npix) {
  __shared__ __attribute__((aligned(32))) unsigned short lB[2][TKC * TNB];
  const int  tid  = threadIdx.x;
  const int  wave = tid >> 5;
  const int  lane = tid & 31;
  const bool lo   = lane < 16;
  const int  mt   = blockIdx.y * 2;           // two 16-row A tiles per block
  const long nblock = (long)blockIdx.x * TNB;

  const unsigned short* aptr0 = Apack + (size_t)mt * nch * 512 + (size_t)lane * 16;
  const unsigned short* aptr1 = aptr0 + (size_t)nch * 512;
  v8f acc00 = {}, acc01 = {}, acc10 = {}, acc11 = {};

  auto compute = [&](int cur, int c) {
    v16us au0 = *(const v16us*)(aptr0 + (size_t)c * 512);        // 2 x global b128
    v16us au1 = *(const v16us*)(aptr1 + (size_t)c * 512);        // 2 x global b128
    const unsigned short* lbb = &lB[cur][0];
    int n0 = wave * 32 + (lane & 15);
    v16us bu0 = *(const v16us*)(lbb + n0 * 32 + (lo ? 0 : 16));  // 2 x ds b128
    v16us bu1 = *(const v16us*)(lbb + (n0 + 16) * 32 + (lo ? 0 : 16));
    v16bf av0 = __builtin_bit_cast(v16bf, au0);
    v16bf av1 = __builtin_bit_cast(v16bf, au1);
    v16bf bv0 = __builtin_bit_cast(v16bf, bu0);
    v16bf bv1 = __builtin_bit_cast(v16bf, bu1);
    acc00 = __builtin_amdgcn_wmma_f32_16x16x32_bf16(false, av0, false, bv0,
                                                    (short)0, acc00, false, false);
    acc01 = __builtin_amdgcn_wmma_f32_16x16x32_bf16(false, av0, false, bv1,
                                                    (short)0, acc01, false, false);
    acc10 = __builtin_amdgcn_wmma_f32_16x16x32_bf16(false, av1, false, bv0,
                                                    (short)0, acc10, false, false);
    acc11 = __builtin_amdgcn_wmma_f32_16x16x32_bf16(false, av1, false, bv1,
                                                    (short)0, acc11, false, false);
    if (c + 1 < nch) {
      __builtin_prefetch(aptr0 + (size_t)(c + 1) * 512, 0, 2);
      __builtin_prefetch(aptr1 + (size_t)(c + 1) * 512, 0, 2);
    }
  };

#if USE_TDM
  if (wave == 0)
    tdm_issue_b(Bpack + (size_t)nblock * 32, (unsigned int)(uintptr_t)&lB[0][0]);
  for (int c = 0; c < nch; ++c) {
    int cur = c & 1;
    if (wave == 0) {
      if (c + 1 < nch) {
        tdm_issue_b(Bpack + ((size_t)(c + 1) * Npix + nblock) * 32,
                    (unsigned int)(uintptr_t)&lB[cur ^ 1][0]);
        __builtin_amdgcn_s_wait_tensorcnt(1);   // chunk c landed (in-order)
      } else {
        __builtin_amdgcn_s_wait_tensorcnt(0);
      }
    }
    __syncthreads();           // tile c visible to all waves
    compute(cur, c);
    __syncthreads();           // all waves done before buffer is re-filled
  }
#else
  for (int c = 0; c < nch; ++c) {
    __syncthreads();
    for (int i = tid; i < TKC * TNB; i += 128)
      lB[0][i] = Bpack[((size_t)c * Npix + nblock) * 32 + i];
    __syncthreads();
    compute(0, c);
  }
#endif

  // store: f32 16x16 C layout (VGPR r -> M = r + (lane<16 ? 0 : 8)) + bias
  const long ncol = nblock + wave * 32 + (lane & 15);
#pragma unroll
  for (int r = 0; r < 8; ++r) {
    int mg0 = mt * 16 + r + (lo ? 0 : 8);
    int mg1 = mg0 + 16;
    float bb0 = bias[mg0];
    float bb1 = bias[mg1];
    C[(size_t)mg0 * Npix + ncol]      = acc00[r] + bb0;
    C[(size_t)mg0 * Npix + ncol + 16] = acc01[r] + bb0;
    C[(size_t)mg1 * Npix + ncol]      = acc10[r] + bb1;
    C[(size_t)mg1 * Npix + ncol + 16] = acc11[r] + bb1;
  }
}

// ---------------------------------------------------------------------------
// BatchNorm statistics: one block (256 threads = 8 waves) per channel
// ---------------------------------------------------------------------------
__global__ __launch_bounds__(256) void bn_stats_kernel(const float* __restrict__ xc, int Npix,
                                                       float* __restrict__ mean,
                                                       float* __restrict__ inv) {
  int c = blockIdx.x;
  const float* p = xc + (size_t)c * Npix;
  float s = 0.f, s2 = 0.f;
  for (int i = threadIdx.x; i < Npix; i += blockDim.x) {
    float v = p[i];
    s += v; s2 += v * v;
  }
  for (int off = 16; off; off >>= 1) {
    s  += __shfl_down(s,  off, 32);
    s2 += __shfl_down(s2, off, 32);
  }
  __shared__ float ls[8], ls2[8];
  int wave = threadIdx.x >> 5, lane = threadIdx.x & 31;
  if (lane == 0) { ls[wave] = s; ls2[wave] = s2; }
  __syncthreads();
  if (threadIdx.x == 0) {
    float S = 0.f, S2 = 0.f;
    for (int w = 0; w < 8; ++w) { S += ls[w]; S2 += ls2[w]; }
    float m = S / (float)Npix;
    float v = S2 / (float)Npix - m * m;
    mean[c] = m;
    inv[c]  = rsqrtf(v + 1e-5f);
  }
}

// ---------------------------------------------------------------------------
// BN apply + LeakyReLU (+ optional skip add)
// ---------------------------------------------------------------------------
__global__ void bn_apply_kernel(const float* __restrict__ xc,
                                const float* __restrict__ mean, const float* __restrict__ inv,
                                const float* __restrict__ g, const float* __restrict__ be,
                                const float* __restrict__ skip, float* __restrict__ dest,
                                int Npix, int Cout, float slope) {
  long total = (long)Cout * Npix;
  for (long i = (long)blockIdx.x * blockDim.x + threadIdx.x; i < total;
       i += (long)gridDim.x * blockDim.x) {
    int c = (int)(i / Npix);
    float v = (xc[i] - mean[c]) * inv[c] * g[c] + be[c];
    v = v > 0.f ? v : slope * v;
    if (skip) v += skip[i];
    dest[i] = v;
  }
}

// ---------------------------------------------------------------------------
// Bilinear 2x upsample (align_corners)
// ---------------------------------------------------------------------------
__global__ void upsample2x_kernel(const float* __restrict__ x, float* __restrict__ y,
                                  int C, int H) {
  int  Ho    = H * 2;
  long total = (long)C * Ho * Ho;
  float scale = (float)(H - 1) / (float)(Ho - 1);
  for (long i = (long)blockIdx.x * blockDim.x + threadIdx.x; i < total;
       i += (long)gridDim.x * blockDim.x) {
    int xo = (int)(i % Ho);
    int yo = (int)((i / Ho) % Ho);
    int c  = (int)(i / ((long)Ho * Ho));
    float fy = yo * scale, fx = xo * scale;
    int y0 = (int)fy; int y1 = (y0 + 1 < H) ? y0 + 1 : H - 1;
    int x0 = (int)fx; int x1 = (x0 + 1 < H) ? x0 + 1 : H - 1;
    float wy = fy - (float)y0, wx = fx - (float)x0;
    const float* pc = x + (long)c * H * H;
    float a = pc[(long)y0 * H + x0], b = pc[(long)y0 * H + x1];
    float cq = pc[(long)y1 * H + x0], d = pc[(long)y1 * H + x1];
    float top = a + (b - a) * wx;
    float bot = cq + (d - cq) * wx;
    y[i] = top + (bot - top) * wy;
  }
}

// ---------------------------------------------------------------------------
// KNN softmax tail: one wave per pixel j, lane = channel (C == warpSize == 32)
// ---------------------------------------------------------------------------
__global__ __launch_bounds__(256) void knn_kernel(
    const float* __restrict__ res,      // [32][256][256]
    const float* __restrict__ xnoise,   // [32][255*255]
    const int*   __restrict__ nbr,      // [J][48]
    float* __restrict__ Wout,           // [J][48]
    float* __restrict__ Oout,           // [32][J]
    int J, int h, int Hres) {
  __shared__ float Dk[8][48];
  __shared__ float Wk[8][48];
  int wave = threadIdx.x >> 5;
  int lane = threadIdx.x & 31;
  long j = (long)blockIdx.x * 8 + wave;
  if (j >= J) return;
  const long chs = (long)Hres * Hres;
  int yj = (int)(j / h), xj = (int)(j % h);
  float uj = res[(long)lane * chs + (long)yj * Hres + xj];
  for (int k = 0; k < 48; ++k) {
    int n  = nbr[j * 48 + k];
    int yn = n / h, xn = n % h;
    float un = res[(long)lane * chs + (long)yn * Hres + xn];
    float d = uj - un;
    d *= d;
    for (int off = 16; off; off >>= 1) d += __shfl_xor(d, off, 32);
    if (lane == 0) Dk[wave][k] = -sqrtf(d * (1.0f / 32.0f) + 1e-9f);
  }
  __builtin_amdgcn_wave_barrier();
  if (lane == 0) {
    float mx = -3.4e38f;
    for (int k = 0; k < 48; ++k) mx = fmaxf(mx, Dk[wave][k]);
    float s = 0.f;
    for (int k = 0; k < 48; ++k) { float e = expf(Dk[wave][k] - mx); Wk[wave][k] = e; s += e; }
    float is = 1.0f / s;
    for (int k = 0; k < 48; ++k) {
      float wv = Wk[wave][k] * is;
      Wk[wave][k] = wv;
      Wout[j * 48 + k] = wv;
    }
  }
  __builtin_amdgcn_wave_barrier();
  float acc = 0.f;
  for (int k = 0; k < 48; ++k) {
    int n = nbr[j * 48 + k];
    acc += Wk[wave][k] * xnoise[(long)lane * J + n];
  }
  Oout[(long)lane * J + j] = acc;
}

// ---------------------------------------------------------------------------
// Host orchestration
// ---------------------------------------------------------------------------
static void run_conv(hipStream_t stream,
                     const float* in, int Cin, int Hin, int stride, int ks,
                     const float* w, const float* b, const float* g, const float* be,
                     int Cout, float* dest, float slope, const float* skip,
                     unsigned short* wbf, unsigned short* col, float* convout,
                     float* mean, float* inv) {
  int pad = (ks - 1) / 2;
  int Ho  = (Hin + 2 * pad - ks) / stride + 1;
  long Npix = (long)Ho * Ho;
  int K   = Cin * ks * ks;
  int Kp  = (K + 31) & ~31;
  int nch = Kp / 32;

  { long total = (long)(Cout >> 4) * nch * 512;
    int gsz = (int)((total + 255) / 256); if (gsz > 8192) gsz = 8192;
    wcvt_pack_kernel<<<gsz, 256, 0, stream>>>(w, wbf, Cout, K, nch); }

  { long total = (long)nch * Npix * 32;
    int gsz = (int)((total + 255) / 256); if (gsz > 16384) gsz = 16384;
    im2col_pack_kernel<<<gsz, 256, 0, stream>>>(in, col, Cin, Hin, Ho, stride, ks, pad,
                                                nch, (int)Npix); }

  dim3 gdim((unsigned)(Npix / TNB), (unsigned)(Cout >> 5), 1);
  gemm_wmma_bias<<<gdim, 128, 0, stream>>>(wbf, col, b, convout, Cout, nch, (int)Npix);

  bn_stats_kernel<<<Cout, 256, 0, stream>>>(convout, (int)Npix, mean, inv);

  { long total = (long)Cout * Npix;
    int gsz = (int)((total + 255) / 256); if (gsz > 8192) gsz = 8192;
    bn_apply_kernel<<<gsz, 256, 0, stream>>>(convout, mean, inv, g, be, skip, dest,
                                             (int)Npix, Cout, slope); }
}

static void run_upsample(hipStream_t stream, const float* in, float* out, int C, int H) {
  long total = (long)C * (2 * H) * (2 * H);
  int gsz = (int)((total + 255) / 256); if (gsz > 8192) gsz = 8192;
  upsample2x_kernel<<<gsz, 256, 0, stream>>>(in, out, C, H);
}

extern "C" void kernel_launch(void* const* d_in, const int* in_sizes, int n_in,
                              void* d_out, int out_size, void* d_ws, size_t ws_size,
                              hipStream_t stream) {
  (void)in_sizes; (void)n_in; (void)out_size; (void)ws_size;

  const float* x       = (const float*)d_in[0];
  const float* x_noise = (const float*)d_in[1];
  const int*   Nidx    = (const int*)d_in[2];

  // params flattened in dict insertion order, 4-tuples (w,b,g,be):
  // inc1a=0 inc1b=4 down1=8 inc2a=12 inc2b=16 down2=20 inc3a=24 inc3b=28
  // down3=32 inc4a=36 inc4b=40 up1=44 inc5a=48 inc5b=52 up2=56 inc6a=60
  // inc6b=64 up3=68 inc7a=72 inc7b=76 out=80
#define PP(i) ((const float*)d_in[3 + (i)])

  // workspace bump allocator (offsets identical every call -> deterministic)
  char* wp = (char*)d_ws;
  auto alloc = [&](size_t bytes) -> void* {
    void* r = (void*)wp;
    wp += (bytes + 255) & ~(size_t)255;
    return r;
  };
  float* x1   = (float*)alloc((size_t)32  * 65536 * 4);  // skip @256
  float* x2   = (float*)alloc((size_t)64  * 16384 * 4);  // skip @128
  float* x3   = (float*)alloc((size_t)128 * 4096  * 4);  // skip @64
  float* bufA = (float*)alloc((size_t)32  * 65536 * 4);  // generic activations
  float* bufU = (float*)alloc((size_t)64  * 65536 * 4);  // upsample scratch
  float* cvo  = (float*)alloc((size_t)32  * 65536 * 4);  // raw conv output
  unsigned short* wbf = (unsigned short*)alloc((size_t)256 * 2304  * 2);
  unsigned short* col = (unsigned short*)alloc((size_t)288 * 65536 * 2);
  float* mean = (float*)alloc(256 * 4);
  float* inv  = (float*)alloc(256 * 4);

#define CONV(in_, Cin_, H_, s_, ks_, base_, Cout_, dst_, slope_, skip_) \
  run_conv(stream, in_, Cin_, H_, s_, ks_, PP(base_ + 0), PP(base_ + 1), \
           PP(base_ + 2), PP(base_ + 3), Cout_, dst_, slope_, skip_,     \
           wbf, col, cvo, mean, inv)

  const float L = 0.01f;
  // encoder
  CONV(x,    1,   256, 1, 3, 0,  32,  bufA, L, nullptr);   // inc1 c1
  CONV(bufA, 32,  256, 1, 3, 4,  32,  x1,   L, nullptr);   // inc1 c2  -> x1
  CONV(x1,   32,  256, 2, 3, 8,  32,  bufA, L, nullptr);   // down1
  CONV(bufA, 32,  128, 1, 3, 12, 64,  bufA, L, nullptr);   // inc2 c1
  CONV(bufA, 64,  128, 1, 3, 16, 64,  x2,   L, nullptr);   // inc2 c2  -> x2
  CONV(x2,   64,  128, 2, 3, 20, 64,  bufA, L, nullptr);   // down2
  CONV(bufA, 64,  64,  1, 3, 24, 128, bufA, L, nullptr);   // inc3 c1
  CONV(bufA, 128, 64,  1, 3, 28, 128, x3,   L, nullptr);   // inc3 c2  -> x3
  CONV(x3,   128, 64,  2, 3, 32, 128, bufA, L, nullptr);   // down3
  CONV(bufA, 128, 32,  1, 3, 36, 256, bufA, L, nullptr);   // inc4 c1
  CONV(bufA, 256, 32,  1, 3, 40, 256, bufA, L, nullptr);   // inc4 c2  -> x4
  // decoder
  run_upsample(stream, bufA, bufU, 256, 32);
  CONV(bufU, 256, 64,  1, 1, 44, 128, bufA, L, x3);        // up1 + skip x3
  CONV(bufA, 128, 64,  1, 3, 48, 128, bufA, L, nullptr);   // inc5 c1
  CONV(bufA, 128, 64,  1, 3, 52, 128, bufA, L, nullptr);   // inc5 c2  -> x5
  run_upsample(stream, bufA, bufU, 128, 64);
  CONV(bufU, 128, 128, 1, 1, 56, 64,  bufA, L, x2);        // up2 + skip x2
  CONV(bufA, 64,  128, 1, 3, 60, 64,  bufA, L, nullptr);   // inc6 c1
  CONV(bufA, 64,  128, 1, 3, 64, 64,  bufA, L, nullptr);   // inc6 c2  -> x6
  run_upsample(stream, bufA, bufU, 64, 128);
  CONV(bufU, 64,  256, 1, 1, 68, 32,  bufA, L, x1);        // up3 + skip x1
  CONV(bufA, 32,  256, 1, 3, 72, 32,  bufA, L, nullptr);   // inc7 c1
  CONV(bufA, 32,  256, 1, 3, 76, 32,  bufA, L, nullptr);   // inc7 c2  -> x7
  CONV(bufA, 32,  256, 1, 3, 80, 32,  bufA, 0.0f, nullptr);// out conv (ReLU)

  // KNN softmax tail -> d_out = [W (J*48) | out (32*J)]
  const int J = 255 * 255;
  float* Wout = (float*)d_out;
  float* Oout = Wout + (size_t)J * 48;
  int gsz = (J + 7) / 8;
  knn_kernel<<<gsz, 256, 0, stream>>>(bufA, x_noise, Nidx, Wout, Oout, J, 255, 256);

#undef CONV
#undef PP
}
